// GNet_65764539236973
// MI455X (gfx1250) — matrix operand order repaired and checked by reading, not compiled, not run on hardware
//
#include <hip/hip_runtime.h>

// ---------------------------------------------------------------------------
// MI455X (gfx1250) fused GNet forward.
// All GEMMs via v_wmma_f32_16x16x32_f16 (f16 in, f32 accumulate).
// Weights pre-transposed to [Nout][Kpad] f16 so B fragments are two 16B
// loads per lane. Intermediate LDS activation tiles use a row-pair-packed
// layout (dword (row>>1)*W + col, row&1 selects half) so the C/D epilogue is
// v_cvt_pk_rtz + 4x ds_store_b32 per tile, and A-fragment loads are
// 4x ds_load_b128 + 8x v_perm_b32 (done once per layer, register-cached).
// segment_max implemented with global_atomic_max_u32 on ReLU'd (>=0) floats.
// ---------------------------------------------------------------------------

typedef __attribute__((ext_vector_type(16))) _Float16 v16h;
typedef __attribute__((ext_vector_type(8)))  float    v8f;

#define N_DET   20000
#define E_PAIR  1000000
#define NBLK    8
#define WAVES   4   // waves (tiles) per workgroup

// ---- WMMA fragment helpers -------------------------------------------------

// A fragment (16x32 f16, MxK) from row-major f16 [16][stride] starting at base.
// lane l: m = l&15, h = l>>4 ; elems 0..7  <- K = kbase+8h .. +8
//                              elems 8..15 <- K = kbase+16+8h .. +8
__device__ __forceinline__ v16h load_a_frag(const _Float16* base, int stride,
                                            int lane, int kbase) {
  int m = lane & 15;
  int h = (lane >> 4) & 1;
  const _Float16* p = base + (size_t)m * stride + kbase + 8 * h;
  v16h a;
  ((uint4*)&a)[0] = *(const uint4*)p;
  ((uint4*)&a)[1] = *(const uint4*)(p + 16);
  return a;
}

// A fragment from row-pair-packed LDS tile: element (row,col) is the
// (row&1) half of dword (row>>1)*width + col.
__device__ __forceinline__ v16h load_a_frag_pk(const _Float16* base, int width,
                                               int lane, int kbase) {
  int m = lane & 15;
  int h = (lane >> 4) & 1;
  const unsigned* b32 = (const unsigned*)base;
  int p = (m >> 1) * width;
  unsigned sel = (m & 1) ? 0x07060302u : 0x05040100u;
  unsigned d[8], e[8];
  *(uint4*)&d[0] = *(const uint4*)(b32 + p + kbase + 8 * h);
  *(uint4*)&d[4] = *(const uint4*)(b32 + p + kbase + 8 * h + 4);
  *(uint4*)&e[0] = *(const uint4*)(b32 + p + kbase + 16 + 8 * h);
  *(uint4*)&e[4] = *(const uint4*)(b32 + p + kbase + 16 + 8 * h + 4);
  v16h a;
  unsigned* out = (unsigned*)&a;
#pragma unroll
  for (int j = 0; j < 4; ++j) {
    out[j]     = __builtin_amdgcn_perm(d[2 * j + 1], d[2 * j], sel);
    out[4 + j] = __builtin_amdgcn_perm(e[2 * j + 1], e[2 * j], sel);
  }
  return a;
}

// B fragment (32x16 f16, KxN) from transposed weights W[Nout][Kpad].
// lane l: n = l&15, elems 0..7 <- K = kt*32+8h.., elems 8..15 <- K = kt*32+16+8h..
__device__ __forceinline__ v16h load_b_frag(const _Float16* wT, int Kpad,
                                            int lane, int nt, int kt) {
  const _Float16* p = wT + (size_t)(nt * 16 + (lane & 15)) * Kpad
                         + kt * 32 + ((lane >> 4) & 1) * 8;
  v16h b;
  ((uint4*)&b)[0] = *(const uint4*)p;
  ((uint4*)&b)[1] = *(const uint4*)(p + 16);
  return b;
}

__device__ __forceinline__ v8f wmma16(v16h a, v16h b, v8f c) {
  // emits v_wmma_f32_16x16x32_f16
  return __builtin_amdgcn_wmma_f32_16x16x32_f16(false, a, false, b,
                                                (short)0, c, false, false);
}

// acc(+bias) -> relu -> f16 into row-major LDS tile [16][width] (scattered).
__device__ __forceinline__ void store_relu_lds(v8f acc, const float* bias,
                                               _Float16* dst, int width,
                                               int lane, int nt) {
  int m = lane & 15, h = (lane >> 4) & 1;
  int n = nt * 16 + m;
  float bb = bias[n];
#pragma unroll
  for (int r = 0; r < 8; ++r) {
    float v = acc[r] + bb;
    v = v > 0.f ? v : 0.f;
    dst[(r + 8 * h) * width + n] = (_Float16)v;
  }
}

// acc(+bias) -> relu -> row-pair-packed LDS tile: cvt_pk + 4x b32 stores.
__device__ __forceinline__ void store_relu_lds_pk(v8f acc, const float* bias,
                                                  _Float16* dst, int width,
                                                  int lane, int nt) {
  int m = lane & 15, h = (lane >> 4) & 1;
  int n = nt * 16 + m;
  float bb = bias[n];
  unsigned* d32 = (unsigned*)dst;
#pragma unroll
  for (int j = 0; j < 4; ++j) {
    float v0 = acc[2 * j]     + bb; v0 = v0 > 0.f ? v0 : 0.f;
    float v1 = acc[2 * j + 1] + bb; v1 = v1 > 0.f ? v1 : 0.f;
    auto pk = __builtin_amdgcn_cvt_pkrtz(v0, v1);  // v_cvt_pk_rtz_f16_f32
    unsigned u;
    __builtin_memcpy(&u, &pk, 4);
    d32[(4 * h + j) * width + n] = u;
  }
}

// ---- weight prep: convert f32 [B][K][N] -> f16 transposed [B][N][Kpad] -----
__global__ void wprep_kernel(const float* __restrict__ src,
                             _Float16* __restrict__ dst,
                             int B, int K, int Nn, int Kpad) {
  int idx = blockIdx.x * blockDim.x + threadIdx.x;
  int total = B * Nn * Kpad;
  if (idx >= total) return;
  int k = idx % Kpad;
  int n = (idx / Kpad) % Nn;
  int b = idx / (Kpad * Nn);
  float v = (k < K) ? src[((size_t)b * K + k) * Nn + n] : 0.f;
  dst[idx] = (_Float16)v;
}

// ---- det init: f32 copy + f16 shadow ---------------------------------------
__global__ void detinit_kernel(const float* __restrict__ src,
                               float* __restrict__ det,
                               _Float16* __restrict__ deth, int total) {
  int i = blockIdx.x * blockDim.x + threadIdx.x;
  if (i < total) {
    float v = src[i];
    det[i]  = v;
    deth[i] = (_Float16)v;
  }
}

// ---- kernel A: pair MLP 9->256->256->32 (relu each), fused per 16-row tile -
__global__ __launch_bounds__(32 * WAVES)
void pairgen_kernel(const float* __restrict__ pairRaw,
                    const float* __restrict__ b0, const float* __restrict__ b1,
                    const float* __restrict__ b2,
                    const _Float16* __restrict__ W0t,  // [256][32]
                    const _Float16* __restrict__ W1t,  // [256][256]
                    const _Float16* __restrict__ W2t,  // [32][256]
                    _Float16* __restrict__ pairFeat,   // [E][32]
                    int numTiles) {
  __shared__ _Float16 lds[WAVES][16 * 32 + 16 * 256 + 16 * 256];
  int wave = threadIdx.x >> 5, lane = threadIdx.x & 31;
  int tile = blockIdx.x * WAVES + wave;
  if (tile >= numTiles) return;                 // wave-uniform
  _Float16* A0 = &lds[wave][0];
  _Float16* H1 = A0 + 16 * 32;    // row-pair packed [8][256] dwords
  _Float16* H2 = H1 + 16 * 256;   // row-pair packed
  int r0 = tile * 16;

  // stage padded input tile [16][32] row-major, cols 9..31 zero
  for (int i = lane; i < 16 * 32; i += 32) A0[i] = (_Float16)0.f;
  for (int i = lane; i < 16 * 9; i += 32) {
    int r = i / 9, c = i % 9;
    A0[r * 32 + c] = (_Float16)pairRaw[(size_t)(r0 + r) * 9 + c];
  }
  // same-wave DS ops are in-order; no barrier needed

  // layer0: K=32(pad of 9), N=256
  v16h a0 = load_a_frag(A0, 32, lane, 0);
  for (int nt = 0; nt < 16; ++nt) {
    v8f acc = {};
    acc = wmma16(a0, load_b_frag(W0t, 32, lane, nt, 0), acc);
    store_relu_lds_pk(acc, b0, H1, 256, lane, nt);
  }
  // layer1: K=256, N=256 -- A fragments cached once in registers
  v16h af1[8];
#pragma unroll
  for (int kt = 0; kt < 8; ++kt) af1[kt] = load_a_frag_pk(H1, 256, lane, kt * 32);
  for (int nt = 0; nt < 16; ++nt) {
    v8f acc = {};
#pragma unroll
    for (int kt = 0; kt < 8; ++kt)
      acc = wmma16(af1[kt], load_b_frag(W1t, 256, lane, nt, kt), acc);
    store_relu_lds_pk(acc, b1, H2, 256, lane, nt);
  }
  // layer2: K=256, N=32 -> global f16 (row-major)
  v16h af2[8];
#pragma unroll
  for (int kt = 0; kt < 8; ++kt) af2[kt] = load_a_frag_pk(H2, 256, lane, kt * 32);
  int m = lane & 15, h = (lane >> 4) & 1;
  for (int nt = 0; nt < 2; ++nt) {
    v8f acc = {};
#pragma unroll
    for (int kt = 0; kt < 8; ++kt)
      acc = wmma16(af2[kt], load_b_frag(W2t, 256, lane, nt, kt), acc);
    int n = nt * 16 + m;
    float bb = b2[n];
#pragma unroll
    for (int r = 0; r < 8; ++r) {
      float v = acc[r] + bb;
      v = v > 0.f ? v : 0.f;
      pairFeat[(size_t)(r0 + r + 8 * h) * 32 + n] = (_Float16)v;
    }
  }
}

// ---- kernel B: f1 = relu(det @ fc1_W[i] + b)  [N,128]x[128,32] -------------
__global__ __launch_bounds__(32 * WAVES)
void f1_kernel(const _Float16* __restrict__ deth,
               const _Float16* __restrict__ WfcT,  // [32][128]
               const float* __restrict__ bfc,
               _Float16* __restrict__ f1h, int numTiles) {
  int wave = threadIdx.x >> 5, lane = threadIdx.x & 31;
  int tile = blockIdx.x * WAVES + wave;
  if (tile >= numTiles) return;
  int r0 = tile * 16;
  const _Float16* A = deth + (size_t)r0 * 128;
  int m = lane & 15, h = (lane >> 4) & 1;
  v16h af[4];
#pragma unroll
  for (int kt = 0; kt < 4; ++kt) af[kt] = load_a_frag(A, 128, lane, kt * 32);
  for (int nt = 0; nt < 2; ++nt) {
    v8f acc = {};
#pragma unroll
    for (int kt = 0; kt < 4; ++kt)
      acc = wmma16(af[kt], load_b_frag(WfcT, 128, lane, nt, kt), acc);
    int n = nt * 16 + m;
    float bb = bfc[n];
#pragma unroll
    for (int r = 0; r < 8; ++r) {
      float v = acc[r] + bb;
      v = v > 0.f ? v : 0.f;
      f1h[(size_t)(r0 + r + 8 * h) * 32 + n] = (_Float16)v;
    }
  }
}

// ---- kernel C: comb=[pairFeat|cF|nF] -> pw1 -> pw2 -> atomic segment max ---
__global__ __launch_bounds__(32 * WAVES)
void pairblock_kernel(const _Float16* __restrict__ pairFeat,
                      const _Float16* __restrict__ f1h,
                      const int* __restrict__ cIdx,
                      const int* __restrict__ nIdx,
                      const _Float16* __restrict__ Wpw1T,  // [64][96]
                      const float* __restrict__ bpw1,
                      const _Float16* __restrict__ Wpw2T,  // [64][64]
                      const float* __restrict__ bpw2,
                      unsigned int* __restrict__ pooledBits,  // [N][64] f32 bits
                      int numTiles) {
  __shared__ _Float16 lds[WAVES][16 * 96 + 16 * 64];
  int wave = threadIdx.x >> 5, lane = threadIdx.x & 31;
  int tile = blockIdx.x * WAVES + wave;
  if (tile >= numTiles) return;
  _Float16* C = &lds[wave][0];       // row-major [16][96]
  _Float16* H = C + 16 * 96;         // row-pair packed
  int r0 = tile * 16;
  int m = lane & 15, half = (lane >> 4) & 1;
  int rr = r0 + m;
  int ci = cIdx[rr], ni = nIdx[rr];

  // per-lane cached segment ids for the 8 rows this lane's C/D elems hit
  int segs[8];
#pragma unroll
  for (int r = 0; r < 8; ++r) segs[r] = cIdx[r0 + r + 8 * half];

  if (half == 0) {
    const uint4* ps = (const uint4*)(pairFeat + (size_t)rr * 32);
    uint4* pd = (uint4*)(C + m * 96);
    pd[0] = ps[0]; pd[1] = ps[1]; pd[2] = ps[2]; pd[3] = ps[3];
    const uint4* cs = (const uint4*)(f1h + (size_t)ci * 32);
    uint4* cd = (uint4*)(C + m * 96 + 32);
    cd[0] = cs[0]; cd[1] = cs[1]; cd[2] = cs[2]; cd[3] = cs[3];
  } else {
    uint4* nd = (uint4*)(C + m * 96 + 64);
    if (ci == ni) {
      uint4 z = {0u, 0u, 0u, 0u};
      nd[0] = z; nd[1] = z; nd[2] = z; nd[3] = z;
    } else {
      const uint4* ns = (const uint4*)(f1h + (size_t)ni * 32);
      nd[0] = ns[0]; nd[1] = ns[1]; nd[2] = ns[2]; nd[3] = ns[3];
    }
  }
  // same-wave DS ordering covers fill -> fragment reads

  // pw1: K=96, N=64
  v16h afc[3];
#pragma unroll
  for (int kt = 0; kt < 3; ++kt) afc[kt] = load_a_frag(C, 96, lane, kt * 32);
  for (int nt = 0; nt < 4; ++nt) {
    v8f acc = {};
#pragma unroll
    for (int kt = 0; kt < 3; ++kt)
      acc = wmma16(afc[kt], load_b_frag(Wpw1T, 96, lane, nt, kt), acc);
    store_relu_lds_pk(acc, bpw1, H, 64, lane, nt);
  }
  // pw2: K=64, N=64, then relu + atomic max into pooled (values >= 0)
  v16h afh[2];
#pragma unroll
  for (int kt = 0; kt < 2; ++kt) afh[kt] = load_a_frag_pk(H, 64, lane, kt * 32);
  for (int nt = 0; nt < 4; ++nt) {
    v8f acc = {};
#pragma unroll
    for (int kt = 0; kt < 2; ++kt)
      acc = wmma16(afh[kt], load_b_frag(Wpw2T, 64, lane, nt, kt), acc);
    int n = nt * 16 + m;
    float bb = bpw2[n];
#pragma unroll
    for (int r = 0; r < 8; ++r) {
      float v = acc[r] + bb;
      v = v > 0.f ? v : 0.f;
      atomicMax(pooledBits + (size_t)segs[r] * 64 + n, __float_as_uint(v));
    }
  }
}

// ---- kernel D: pooled -> pm1 -> pm2 -> out + residual relu into det --------
__global__ __launch_bounds__(32 * WAVES)
void refine_kernel(const float* __restrict__ pooled,  // [N][64]
                   const _Float16* __restrict__ Wpm1T, const float* __restrict__ bpm1,
                   const _Float16* __restrict__ Wpm2T, const float* __restrict__ bpm2,
                   const _Float16* __restrict__ WoutT,  // [128][64]
                   const float* __restrict__ bout,
                   float* __restrict__ det, _Float16* __restrict__ deth,
                   int numTiles) {
  __shared__ _Float16 lds[WAVES][16 * 64 * 3];
  int wave = threadIdx.x >> 5, lane = threadIdx.x & 31;
  int tile = blockIdx.x * WAVES + wave;
  if (tile >= numTiles) return;
  _Float16* P  = &lds[wave][0];     // row-major [16][64]
  _Float16* H1 = P + 16 * 64;       // row-pair packed
  _Float16* H2 = H1 + 16 * 64;      // row-pair packed
  int r0 = tile * 16;
  int m = lane & 15, half = (lane >> 4) & 1;

  for (int i = lane; i < 16 * 64; i += 32) {
    int r = i >> 6, c = i & 63;
    P[i] = (_Float16)pooled[(size_t)(r0 + r) * 64 + c];
  }
  // pm1: 64->64
  v16h afp[2];
#pragma unroll
  for (int kt = 0; kt < 2; ++kt) afp[kt] = load_a_frag(P, 64, lane, kt * 32);
  for (int nt = 0; nt < 4; ++nt) {
    v8f acc = {};
#pragma unroll
    for (int kt = 0; kt < 2; ++kt)
      acc = wmma16(afp[kt], load_b_frag(Wpm1T, 64, lane, nt, kt), acc);
    store_relu_lds_pk(acc, bpm1, H1, 64, lane, nt);
  }
  // pm2: 64->64
  v16h af1[2];
#pragma unroll
  for (int kt = 0; kt < 2; ++kt) af1[kt] = load_a_frag_pk(H1, 64, lane, kt * 32);
  for (int nt = 0; nt < 4; ++nt) {
    v8f acc = {};
#pragma unroll
    for (int kt = 0; kt < 2; ++kt)
      acc = wmma16(af1[kt], load_b_frag(Wpm2T, 64, lane, nt, kt), acc);
    store_relu_lds_pk(acc, bpm2, H2, 64, lane, nt);
  }
  // out: 64->128 (no relu), then det = relu(det + refined)
  v16h af2[2];
#pragma unroll
  for (int kt = 0; kt < 2; ++kt) af2[kt] = load_a_frag_pk(H2, 64, lane, kt * 32);
  for (int nt = 0; nt < 8; ++nt) {
    v8f acc = {};
#pragma unroll
    for (int kt = 0; kt < 2; ++kt)
      acc = wmma16(af2[kt], load_b_frag(WoutT, 64, lane, nt, kt), acc);
    int n = nt * 16 + m;
    float bb = bout[n];
#pragma unroll
    for (int r = 0; r < 8; ++r) {
      size_t idx = (size_t)(r0 + r + 8 * half) * 128 + n;
      float v = det[idx] + acc[r] + bb;
      v = v > 0.f ? v : 0.f;
      det[idx]  = v;
      deth[idx] = (_Float16)v;
    }
  }
}

// ---- kernel E: 3x (128->128 relu) + pred 128->1 ----------------------------
__global__ __launch_bounds__(32 * WAVES)
void score_kernel(const _Float16* __restrict__ deth,
                  const _Float16* __restrict__ WscT,  // [3][128][128]
                  const float* __restrict__ bsc,      // [3][128]
                  const float* __restrict__ predW, const float* __restrict__ predB,
                  float* __restrict__ out, int numTiles) {
  __shared__ _Float16 lds[WAVES][2 * 16 * 128];
  int wave = threadIdx.x >> 5, lane = threadIdx.x & 31;
  int tile = blockIdx.x * WAVES + wave;
  if (tile >= numTiles) return;
  _Float16* Ha = &lds[wave][0];          // layer0 out: row-pair packed
  _Float16* Hb = Ha + 16 * 128;          // layer1 out: row-pair packed
  int r0 = tile * 16;

  {  // layer 0: A from global (row-major)
    v16h af[4];
#pragma unroll
    for (int kt = 0; kt < 4; ++kt)
      af[kt] = load_a_frag(deth + (size_t)r0 * 128, 128, lane, kt * 32);
    for (int nt = 0; nt < 8; ++nt) {
      v8f acc = {};
#pragma unroll
      for (int kt = 0; kt < 4; ++kt)
        acc = wmma16(af[kt], load_b_frag(WscT, 128, lane, nt, kt), acc);
      store_relu_lds_pk(acc, bsc, Ha, 128, lane, nt);
    }
  }
  {  // layer 1
    v16h af[4];
#pragma unroll
    for (int kt = 0; kt < 4; ++kt) af[kt] = load_a_frag_pk(Ha, 128, lane, kt * 32);
    for (int nt = 0; nt < 8; ++nt) {
      v8f acc = {};
#pragma unroll
      for (int kt = 0; kt < 4; ++kt)
        acc = wmma16(af[kt], load_b_frag(WscT + 128 * 128, 128, lane, nt, kt), acc);
      store_relu_lds_pk(acc, bsc + 128, Hb, 128, lane, nt);
    }
  }
  {  // layer 2: store row-major into Ha so pred reads contiguously
    v16h af[4];
#pragma unroll
    for (int kt = 0; kt < 4; ++kt) af[kt] = load_a_frag_pk(Hb, 128, lane, kt * 32);
    for (int nt = 0; nt < 8; ++nt) {
      v8f acc = {};
#pragma unroll
      for (int kt = 0; kt < 4; ++kt)
        acc = wmma16(af[kt], load_b_frag(WscT + 2 * 128 * 128, 128, lane, nt, kt), acc);
      store_relu_lds(acc, bsc + 256, Ha, 128, lane, nt);
    }
  }
  // pred: per-row dot(128) + bias, K split across the two lane halves,
  // reduced with a wave32 xor-shuffle.
  {
    int m = lane & 15, half = (lane >> 4) & 1;
    const _Float16* row = Ha + m * 128 + half * 64;
    const float* w = predW + half * 64;
    float s = 0.f;
#pragma unroll
    for (int k = 0; k < 64; ++k) s += (float)row[k] * w[k];
    s += __shfl_xor(s, 16, 32);
    if (half == 0) out[r0 + m] = s + predB[0];
  }
}

// ---------------------------------------------------------------------------
extern "C" void kernel_launch(void* const* d_in, const int* in_sizes, int n_in,
                              void* d_out, int out_size, void* d_ws, size_t ws_size,
                              hipStream_t stream) {
  (void)in_sizes; (void)n_in; (void)out_size; (void)ws_size;
  const float* detF    = (const float*)d_in[0];
  const float* pairRaw = (const float*)d_in[1];
  const int*   cI      = (const int*)d_in[2];
  const int*   nI      = (const int*)d_in[3];
  const float* g0W = (const float*)d_in[4],  *g0b = (const float*)d_in[5];
  const float* g1W = (const float*)d_in[6],  *g1b = (const float*)d_in[7];
  const float* g2W = (const float*)d_in[8],  *g2b = (const float*)d_in[9];
  const float* fcW = (const float*)d_in[10], *fcb = (const float*)d_in[11];
  const float* p1W = (const float*)d_in[12], *p1b = (const float*)d_in[13];
  const float* p2W = (const float*)d_in[14], *p2b = (const float*)d_in[15];
  const float* m1W = (const float*)d_in[16], *m1b = (const float*)d_in[17];
  const float* m2W = (const float*)d_in[18], *m2b = (const float*)d_in[19];
  const float* oW  = (const float*)d_in[20], *ob  = (const float*)d_in[21];
  const float* sW  = (const float*)d_in[22], *sb  = (const float*)d_in[23];
  const float* pW  = (const float*)d_in[24], *pb  = (const float*)d_in[25];
  float* out = (float*)d_out;

  // workspace layout (256B aligned slices)
  char* ws = (char*)d_ws;
  size_t off = 0;
  auto take = [&](size_t bytes) -> char* {
    char* p = ws + off;
    off = (off + bytes + 255) & ~(size_t)255;
    return p;
  };
  _Float16* W0t  = (_Float16*)take((size_t)256 * 32 * 2);
  _Float16* W1t  = (_Float16*)take((size_t)256 * 256 * 2);
  _Float16* W2t  = (_Float16*)take((size_t)32 * 256 * 2);
  _Float16* Wfct = (_Float16*)take((size_t)NBLK * 32 * 128 * 2);
  _Float16* Wp1t = (_Float16*)take((size_t)NBLK * 64 * 96 * 2);
  _Float16* Wp2t = (_Float16*)take((size_t)NBLK * 64 * 64 * 2);
  _Float16* Wm1t = (_Float16*)take((size_t)NBLK * 64 * 64 * 2);
  _Float16* Wm2t = (_Float16*)take((size_t)NBLK * 64 * 64 * 2);
  _Float16* Wot  = (_Float16*)take((size_t)NBLK * 128 * 64 * 2);
  _Float16* Wst  = (_Float16*)take((size_t)3 * 128 * 128 * 2);
  _Float16* pairFeat = (_Float16*)take((size_t)E_PAIR * 32 * 2);
  _Float16* f1h      = (_Float16*)take((size_t)N_DET * 32 * 2);
  float*    pooled   = (float*)take((size_t)N_DET * 64 * 4);
  float*    det      = (float*)take((size_t)N_DET * 128 * 4);
  _Float16* deth     = (_Float16*)take((size_t)N_DET * 128 * 2);

  auto prep = [&](const float* src, _Float16* dst, int B, int K, int Nn, int Kpad) {
    int total = B * Nn * Kpad;
    wprep_kernel<<<(total + 255) / 256, 256, 0, stream>>>(src, dst, B, K, Nn, Kpad);
  };
  prep(g0W, W0t, 1, 9, 256, 32);
  prep(g1W, W1t, 1, 256, 256, 256);
  prep(g2W, W2t, 1, 256, 32, 256);
  prep(fcW, Wfct, NBLK, 128, 32, 128);
  prep(p1W, Wp1t, NBLK, 96, 64, 96);
  prep(p2W, Wp2t, NBLK, 64, 64, 64);
  prep(m1W, Wm1t, NBLK, 64, 64, 64);
  prep(m2W, Wm2t, NBLK, 64, 64, 64);
  prep(oW,  Wot,  NBLK, 64, 128, 64);
  prep(sW,  Wst,  3, 128, 128, 128);

  {
    int total = N_DET * 128;
    detinit_kernel<<<(total + 255) / 256, 256, 0, stream>>>(detF, det, deth, total);
  }

  const int tilesE = E_PAIR / 16;  // 62500
  const int tilesN = N_DET / 16;   // 1250
  const int gridE  = (tilesE + WAVES - 1) / WAVES;
  const int gridN  = (tilesN + WAVES - 1) / WAVES;

  pairgen_kernel<<<gridE, 32 * WAVES, 0, stream>>>(pairRaw, g0b, g1b, g2b,
                                                   W0t, W1t, W2t, pairFeat, tilesE);

  for (int i = 0; i < NBLK; ++i) {
    f1_kernel<<<gridN, 32 * WAVES, 0, stream>>>(
        deth, Wfct + (size_t)i * 32 * 128, fcb + (size_t)i * 32, f1h, tilesN);
    hipMemsetAsync(pooled, 0, (size_t)N_DET * 64 * 4, stream);
    pairblock_kernel<<<gridE, 32 * WAVES, 0, stream>>>(
        pairFeat, f1h, cI, nI,
        Wp1t + (size_t)i * 64 * 96, p1b + (size_t)i * 64,
        Wp2t + (size_t)i * 64 * 64, p2b + (size_t)i * 64,
        (unsigned int*)pooled, tilesE);
    refine_kernel<<<gridN, 32 * WAVES, 0, stream>>>(
        pooled,
        Wm1t + (size_t)i * 64 * 64, m1b + (size_t)i * 64,
        Wm2t + (size_t)i * 64 * 64, m2b + (size_t)i * 64,
        Wot + (size_t)i * 128 * 64, ob + (size_t)i * 128,
        det, deth, tilesN);
  }

  score_kernel<<<gridN, 32 * WAVES, 0, stream>>>(deth, Wst, sb, pW, pb, out, tilesN);
}
